// PAGPF_38156489458202
// MI455X (gfx1250) — matrix operand order repaired and checked by reading, not compiled, not run on hardware
//
#include <hip/hip_runtime.h>

// ---------------------------------------------------------------------------
// MI455X (gfx1250) implementation of the GCGFN / PolaAttention block.
// All channel-mixing matmuls run on v_wmma_f32_16x16x32_bf16 (wave32 WMMA),
// activations kept token-major [B,N,C]; bf16 activation storage between GEMMs.
// GCGFN depthwise-conv tiles are staged into LDS via the Tensor Data Mover
// (tensor_load_to_lds + s_wait_tensorcnt) when the toolchain exposes it.
// ---------------------------------------------------------------------------

typedef __bf16 bf16_t;
typedef __attribute__((ext_vector_type(16))) __bf16 v16bf;
typedef __attribute__((ext_vector_type(8)))  __bf16 v8bf;
typedef __attribute__((ext_vector_type(8)))  float  v8f;
typedef __attribute__((ext_vector_type(4)))  unsigned int u32x4;
typedef __attribute__((ext_vector_type(8)))  int i32x8;
typedef __attribute__((ext_vector_type(4)))  int i32x4;

#define B_   16
#define N_   4096
#define C_   64
#define BN_  (B_ * N_)
#define H_   4
#define MAXPOS_ 321

#if defined(__has_builtin)
#if __has_builtin(__builtin_amdgcn_tensor_load_to_lds) && \
    __has_builtin(__builtin_amdgcn_s_wait_tensorcnt)
#define USE_TDM 1
#endif
#endif
#ifndef USE_TDM
#define USE_TDM 0
#endif

#if USE_TDM
#pragma message("CDNA5 probe: tensor_load_to_lds builtin FOUND - TDM path enabled")
#else
#pragma message("CDNA5 probe: tensor_load_to_lds builtin MISSING - scalar LDS fill")
#endif
#if defined(__has_builtin)
#if __has_builtin(__builtin_amdgcn_global_load_async_to_lds_b128)
#pragma message("CDNA5 probe: global_load_async_to_lds_b128 builtin FOUND")
#else
#pragma message("CDNA5 probe: global_load_async_to_lds_b128 builtin MISSING")
#endif
#endif

// ---------------------------------------------------------------------------
// fp32 -> bf16 weight conversion / zero-fill
// ---------------------------------------------------------------------------
__global__ void f2bf_kernel(const float* __restrict__ s, bf16_t* __restrict__ d, int n) {
    int i = blockIdx.x * 256 + threadIdx.x;
    if (i < n) d[i] = (bf16_t)s[i];
}

__global__ void zero_f32_kernel(float* __restrict__ p, int n) {
    int i = blockIdx.x * 256 + threadIdx.x;
    if (i < n) p[i] = 0.f;
}

// ---------------------------------------------------------------------------
// Per-token LayerNorm over C=64; one wave32 per token (wave32 __shfl_xor).
// ---------------------------------------------------------------------------
__global__ __launch_bounds__(256)
void ln_c64_kernel(const float* __restrict__ x, const float* __restrict__ w,
                   const float* __restrict__ b, bf16_t* __restrict__ outB,
                   float* __restrict__ outF, float eps, int ntok) {
    int tok = blockIdx.x * 8 + (threadIdx.x >> 5);
    if (tok >= ntok) return;
    int lane = threadIdx.x & 31;
    const float* xr = x + (size_t)tok * C_;
    float v0 = xr[lane], v1 = xr[lane + 32];
    float s = v0 + v1;
    for (int m = 16; m; m >>= 1) s += __shfl_xor(s, m, 32);
    float mu = s * (1.f / 64.f);
    float d0 = v0 - mu, d1 = v1 - mu;
    float q = d0 * d0 + d1 * d1;
    for (int m = 16; m; m >>= 1) q += __shfl_xor(q, m, 32);
    float rinv = rsqrtf(q * (1.f / 64.f) + eps);
    float y0 = d0 * rinv * w[lane] + b[lane];
    float y1 = d1 * rinv * w[lane + 32] + b[lane + 32];
    size_t o = (size_t)tok * C_;
    outB[o + lane]      = (bf16_t)y0;
    outB[o + lane + 32] = (bf16_t)y1;
    if (outF) { outF[o + lane] = y0; outF[o + lane + 32] = y1; }
}

// ---------------------------------------------------------------------------
// BiasNorm over C=64 per token: out = x / rms(x - bias) * exp(gamma)
// ---------------------------------------------------------------------------
__global__ __launch_bounds__(256)
void biasnorm_kernel(const float* __restrict__ x, const float* __restrict__ bias,
                     const float* __restrict__ gamma, float* __restrict__ out, int ntok) {
    int tok = blockIdx.x * 8 + (threadIdx.x >> 5);
    if (tok >= ntok) return;
    int lane = threadIdx.x & 31;
    const float* xr = x + (size_t)tok * C_;
    float v0 = xr[lane], v1 = xr[lane + 32];
    float c0 = v0 - bias[lane], c1 = v1 - bias[lane + 32];
    float q = c0 * c0 + c1 * c1;
    for (int m = 16; m; m >>= 1) q += __shfl_xor(q, m, 32);
    float rinv = rsqrtf(q * (1.f / 64.f) + 1e-5f);
    float g = expf(gamma[0]);
    out[(size_t)tok * C_ + lane]      = v0 * rinv * g;
    out[(size_t)tok * C_ + lane + 32] = v1 * rinv * g;
}

// ---------------------------------------------------------------------------
// WMMA GEMM:  out[M,O] = A[M,K](bf16) * W[O,K]^T(bf16) + bias
// EPI: 0 = store bf16
//      1 = *gate(bf16 aux, ld=256) -> store bf16
//      2 = *scale[col] + resid(ld=64) -> store f32
//      3 = store f32 (no bias)
//      4 = + resid(ld=64) -> store f32
// LDA/LDO are compile-time so row strides fold into immediate offsets
// (no per-store 64-bit address chains).
// 8 waves/block; each wave: one 16-col strip x 4 M-tiles; B fragments kept in
// registers across M-tiles; all A fragments preloaded before the MMA chain so
// global_load_b128s clause and v_wmma issue behind a single wait.
// Fragment layouts per CDNA5 ISA 7.12.2 (wave32).
// ---------------------------------------------------------------------------
template <int KC, int EPI, int LDA, int LDO>   // KC = K/32
__global__ __launch_bounds__(256)
void wmma_gemm_kernel(const bf16_t* __restrict__ A,
                      const bf16_t* __restrict__ W,
                      const float* __restrict__ bias,
                      int M, int O,
                      const bf16_t* __restrict__ mul_aux,   // ld = 256
                      const float* __restrict__ scale,
                      const float* __restrict__ resid,      // ld = 64
                      float* __restrict__ outF, bf16_t* __restrict__ outB) {
    const int lane = threadIdx.x & 31;
    const int wave = threadIdx.x >> 5;
    const int nt_o = O >> 4;
    const int mg   = M >> 6;              // groups of 4 M-tiles
    int grp = blockIdx.x * 8 + wave;
    if (grp >= mg * nt_o) return;
    int mgi = grp / nt_o, nt = grp % nt_o;
    int n0 = nt << 4;

    const int r  = lane & 15;             // A row / B col / D col within tile
    const int hi = lane >> 4;

    // --- preload B fragments (K x 16 strip of W^T), reused by 4 M-tiles ---
    v16bf bw[KC];
    const bf16_t* wrow = W + (size_t)(n0 + r) * (KC * 32) + hi * 8;
#pragma unroll
    for (int i = 0; i < KC; ++i) {
        v8bf blo = *(const v8bf*)(wrow + i * 32);
        v8bf bhi = *(const v8bf*)(wrow + i * 32 + 16);
        bw[i] = __builtin_shufflevector(blo, bhi,
            0, 1, 2, 3, 4, 5, 6, 7, 8, 9, 10, 11, 12, 13, 14, 15);
    }

    int col = n0 + r;
    float bi = (EPI == 3) ? 0.f : bias[col];
    float sc = (EPI == 2) ? scale[col] : 1.f;

#pragma unroll
    for (int s = 0; s < 4; ++s) {
        int m0 = (mgi * 4 + s) << 4;
        const bf16_t* arow = A + (size_t)(m0 + r) * LDA + hi * 8;
        if (s < 3) __builtin_prefetch(arow + 16 * LDA, 0, 3);  // global_prefetch_b8
        // preload all A fragments of this tile, then run the MMA chain
        v16bf aw[KC];
#pragma unroll
        for (int i = 0; i < KC; ++i) {
            v8bf alo = *(const v8bf*)(arow + i * 32);
            v8bf ahi = *(const v8bf*)(arow + i * 32 + 16);
            aw[i] = __builtin_shufflevector(alo, ahi,
                0, 1, 2, 3, 4, 5, 6, 7, 8, 9, 10, 11, 12, 13, 14, 15);
        }
        v8f acc = {};
#pragma unroll
        for (int i = 0; i < KC; ++i)
            acc = __builtin_amdgcn_wmma_f32_16x16x32_bf16(
                false, aw[i], false, bw[i], (short)0, acc, false, false);

        // epilogue: base pointers once, constant row offsets thereafter
        int row0 = m0 + hi * 8;
        const bf16_t* auxp = (EPI == 1) ? mul_aux + (size_t)row0 * 256 + col : nullptr;
        const float*  resp = (EPI == 2 || EPI == 4) ? resid + (size_t)row0 * 64 + col : nullptr;
        float*  ofp = (EPI >= 2) ? outF + (size_t)row0 * LDO + col : nullptr;
        bf16_t* obp = (EPI <= 1) ? outB + (size_t)row0 * LDO + col : nullptr;
#pragma unroll
        for (int i2 = 0; i2 < 8; ++i2) {
            float y = acc[i2] + bi;
            if constexpr (EPI == 1) y *= (float)auxp[i2 * 256];
            if constexpr (EPI == 2) y = y * sc + resp[i2 * 64];
            if constexpr (EPI == 4) y += resp[i2 * 64];
            if constexpr (EPI == 0 || EPI == 1)
                obp[i2 * LDO] = (bf16_t)y;
            else
                ofp[i2 * LDO] = y;
        }
    }
}

// ---------------------------------------------------------------------------
// GCGFN depthwise conv along N (one branch of 64 channels).
// Halo tile staged in LDS — via the Tensor Data Mover on interior blocks
// (2D tile: rows x 64 bf16 out of the [BN,256] activation tensor), scalar
// fill on boundary blocks / toolchains without the TDM builtin.
// Produces both lka-dwconv and x-dwconv outputs (shared input loads).
// ---------------------------------------------------------------------------
__global__ __launch_bounds__(256)
void gcgfn_dwconv_kernel(const bf16_t* __restrict__ a,        // [B*N,256]
                         const float* __restrict__ lw, const float* __restrict__ lb,
                         const float* __restrict__ xw, const float* __restrict__ xb,
                         int ksz, int branch,
                         bf16_t* __restrict__ olka, bf16_t* __restrict__ oxk) {
    const int BLKN = 32;
    __shared__ bf16_t tile[(BLKN + 30) * 64];   // sole LDS alloc -> LDS offset 0
    int pad  = (ksz - 1) >> 1;
    int b    = blockIdx.x / (N_ / BLKN);
    int nch  = blockIdx.x % (N_ / BLKN);
    int nb   = nch * BLKN;
    int rows = BLKN + 2 * pad;

#if USE_TDM
    bool interior = (nb - pad >= 0) && (nb + BLKN + pad <= N_);
    if (interior) {
        if (threadIdx.x < 32) {   // one wave issues the TDM descriptor
            unsigned long long gaddr = (unsigned long long)(size_t)
                (a + ((size_t)b * N_ + (nb - pad)) * 256 + branch * 64);
            // ---- D# group 0: count=1 | lds_addr | global_addr | type=2 ----
            u32x4 g0;
            g0.x = 1u;                                   // count=1, user mode
            g0.y = 0u;                                   // lds_addr (tile @ 0)
            g0.z = (unsigned)(gaddr & 0xffffffffu);
            g0.w = (unsigned)((gaddr >> 32) & 0x1ffffffu) | (2u << 30);
            // ---- D# group 1: data_size=2B, dims/strides/tile ----
            i32x8 g1;
            g1[0] = 1 << 16;                             // data_size=1 (2 bytes)
            g1[1] = 256 << 16;                           // tensor_dim0[15:0]=256
            g1[2] = (BN_ & 0xffff) << 16;                // dim0_hi=0 | dim1_lo
            g1[3] = ((BN_ >> 16) & 0xffff) | (64 << 16); // dim1_hi | tile_dim0=64
            g1[4] = rows & 0xffff;                       // tile_dim1=rows, dim2=0
            g1[5] = 256;                                 // dim0_stride low32
            g1[6] = 0;                                   // dim0_stride hi | dim1_stride lo
            g1[7] = 0;                                   // dim1_stride hi
            i32x4 gz = {0, 0, 0, 0};
#if __clang_major__ >= 23
            i32x8 gz8 = {0, 0, 0, 0, 0, 0, 0, 0};
            __builtin_amdgcn_tensor_load_to_lds(g0, g1, gz, gz, gz8, 0);
#else
            __builtin_amdgcn_tensor_load_to_lds(g0, g1, gz, gz, 0);
#endif
            __builtin_amdgcn_s_wait_tensorcnt(0);
        }
    } else
#endif
    {
        for (int idx = threadIdx.x; idx < rows * 64; idx += 256) {
            int rr = idx >> 6, cc = idx & 63;
            int n = nb + rr - pad;
            bf16_t v = (bf16_t)0.f;
            if (n >= 0 && n < N_) v = a[((size_t)b * N_ + n) * 256 + branch * 64 + cc];
            tile[rr * 64 + cc] = v;
        }
    }
    __syncthreads();

    int c = threadIdx.x & 63;
    for (int ns = threadIdx.x >> 6; ns < BLKN; ns += 4) {
        float aL = lb[c], aX = xb[c];
        for (int t = 0; t < ksz; ++t) {
            float v = (float)tile[(ns + t) * 64 + c];
            aL += lw[c * ksz + t] * v;
            aX += xw[c * ksz + t] * v;
        }
        size_t o = ((size_t)b * N_ + nb + ns) * 256 + branch * 64 + c;
        olka[o] = (bf16_t)aL;
        oxk[o]  = (bf16_t)aX;
    }
}

// ---------------------------------------------------------------------------
// Attention pass A: polynomial features + reduction of kmean / kk^T v.
// Block = 256 tokens of one (b,h). kvred layout per (b,h):
//   [0,32)=sum kk ; [32,288)=kv1[d*8+e] ; [288,544)=kv2[d*8+e]
// ---------------------------------------------------------------------------
__global__ __launch_bounds__(256)
void attn_feat_kernel(const float* __restrict__ qgkv,      // [B*N,256] q|g|k|v
                      const float* __restrict__ scale_p,   // [64]
                      const float* __restrict__ power_p,   // [64]
                      const float* __restrict__ pos,       // [321*64]
                      float* __restrict__ qsim,            // [B*H*N*32]
                      float* __restrict__ kvred) {         // [B*H*544]
    __shared__ float red[256 * 48];
    int bh = blockIdx.x, b = bh >> 2, h = bh & 3;
    int n  = blockIdx.y * 256 + threadIdx.x;
    const float* base = qgkv + ((size_t)b * N_ + n) * 256;
    int pr = n % MAXPOS_;
    float qs[32], kk[32], vv[16];
#pragma unroll
    for (int d = 0; d < 16; ++d) {
        int c = h * 16 + d;
        float sc = log1pf(expf(scale_p[c]));                 // softplus
        float pw = 1.f + 3.f / (1.f + expf(-power_p[c]));    // 1 + ALPHA*sigmoid
        float qv = base[c] / sc;
        float kv = (base[128 + c] + pos[pr * 64 + c]) / sc;
        vv[d] = base[192 + c];
        qs[d]      = qv > 0.f ? powf(qv, pw)  : 0.f;
        qs[16 + d] = qv < 0.f ? powf(-qv, pw) : 0.f;
        kk[d]      = kv > 0.f ? powf(kv, pw)  : 0.f;
        kk[16 + d] = kv < 0.f ? powf(-kv, pw) : 0.f;
    }
    size_t qo = (((size_t)bh * N_) + n) * 32;
#pragma unroll
    for (int i = 0; i < 32; ++i) qsim[qo + i] = qs[i];
    float* row = red + threadIdx.x * 48;
#pragma unroll
    for (int i = 0; i < 32; ++i) row[i] = kk[i];
#pragma unroll
    for (int i = 0; i < 16; ++i) row[32 + i] = vv[i];
    __syncthreads();
    const float innv = 1.f / (float)N_;                       // ninv*ninv
    for (int o = threadIdx.x; o < 544; o += 256) {
        float s = 0.f;
        if (o < 32) {
            for (int t = 0; t < 256; ++t) s += red[t * 48 + o];
        } else if (o < 288) {
            int q = o - 32, d = q >> 3, e = q & 7;
            for (int t = 0; t < 256; ++t) s += red[t * 48 + d] * red[t * 48 + 32 + e];
            s *= innv;
        } else {
            int q = o - 288, d = q >> 3, e = q & 7;
            for (int t = 0; t < 256; ++t) s += red[t * 48 + d] * red[t * 48 + 40 + e];
            s *= innv;
        }
        atomicAdd(&kvred[(size_t)bh * 544 + o], s);
    }
}

// ---------------------------------------------------------------------------
// Attention dwc branch: 5x5 depthwise conv over the torch-reshape view of v.
// vr(b,c16,nn,hh) -> f=((c16*4096+nn)*4+hh): h=f>>16, n=(f&65535)>>4, d=f&15
// output mapped back to [B,N,C] with c64=g>>12, npos=g&4095.
// ---------------------------------------------------------------------------
__global__ __launch_bounds__(256)
void attn_dwc_kernel(const float* __restrict__ qgkv, const float* __restrict__ w,
                     const float* __restrict__ bias, float* __restrict__ vd) {
    int idx = blockIdx.x * 256 + threadIdx.x;        // over B*16*N*4
    if (idx >= B_ * 16 * N_ * 4) return;
    int hh  = idx & 3;
    int nn  = (idx >> 2) & (N_ - 1);
    int c16 = (idx >> 14) & 15;
    int b   = idx >> 18;
    float acc = bias[c16];
    for (int i = 0; i < 5; ++i) {
        int ni = nn + i - 2;
        if (ni < 0 || ni >= N_) continue;
        for (int j = 0; j < 5; ++j) {
            int hj = hh + j - 2;
            if (hj < 0 || hj > 3) continue;
            int f = ((((c16 << 12) | ni) << 2) | hj);
            int h = f >> 16;
            int flat2 = f & 65535;
            int n = flat2 >> 4, d = flat2 & 15;
            acc += w[c16 * 25 + i * 5 + j] *
                   qgkv[((size_t)b * N_ + n) * 256 + 192 + h * 16 + d];
        }
    }
    int g = ((((c16 << 12) | nn) << 2) | hh);
    int c64 = g >> 12, npos = g & (N_ - 1);
    vd[((size_t)b * N_ + npos) * C_ + c64] = acc;
}

// ---------------------------------------------------------------------------
// Attention pass B: z-normalized linear attention output, + vd, * gate,
// stores bf16 A-matrix for the projection GEMM.
// ---------------------------------------------------------------------------
__global__ __launch_bounds__(256)
void attn_out_kernel(const float* __restrict__ qsim, const float* __restrict__ kvred,
                     const float* __restrict__ vd, const float* __restrict__ qgkv,
                     bf16_t* __restrict__ aout) {
    int bh = blockIdx.x, b = bh >> 2, h = bh & 3;
    int n  = blockIdx.y * 256 + threadIdx.x;
    const float* kr = kvred + (size_t)bh * 544;
    float qs[32];
    size_t qo = (((size_t)bh * N_) + n) * 32;
#pragma unroll
    for (int i = 0; i < 32; ++i) qs[i] = qsim[qo + i];
    float d1 = 0.f, d2 = 0.f;
#pragma unroll
    for (int i = 0; i < 32; ++i) {
        float km = kr[i] * (1.f / (float)N_);
        d1 += qs[i] * km;
        d2 += qs[(i + 16) & 31] * km;
    }
    float z1 = 1.f / (d1 + 1e-6f), z2 = 1.f / (d2 + 1e-6f);
    const float* gbase = qgkv + ((size_t)b * N_ + n) * 256 + 64 + h * 16;
    const float* vdb   = vd + ((size_t)b * N_ + n) * C_ + h * 16;
    bf16_t* ob = aout + ((size_t)b * N_ + n) * C_ + h * 16;
    for (int e = 0; e < 8; ++e) {
        float s1 = 0.f, s2 = 0.f;
#pragma unroll
        for (int d = 0; d < 32; ++d) {
            s1 += qs[d] * kr[32 + d * 8 + e];
            s2 += qs[(d + 16) & 31] * kr[288 + d * 8 + e];
        }
        ob[e]     = (bf16_t)((s1 * z1 + vdb[e])     * gbase[e]);
        ob[8 + e] = (bf16_t)((s2 * z2 + vdb[8 + e]) * gbase[8 + e]);
    }
}

// ---------------------------------------------------------------------------
// Host orchestration
// ---------------------------------------------------------------------------
extern "C" void kernel_launch(void* const* d_in, const int* in_sizes, int n_in,
                              void* d_out, int out_size, void* d_ws, size_t ws_size,
                              hipStream_t stream) {
    (void)in_sizes; (void)n_in; (void)out_size; (void)ws_size;

    // --- input indices (setup_inputs dict order, recursively flattened) ---
    // 0:x ; 1..31:g1 ; 32..62:g2 ; 63..71:attn ; 72:norm1_w 73:norm1_b
    // 74:bn_bias 75:bn_gamma
    // g block: 0 ln_w,1 ln_b,2 scale,3 pf_w,4 pf_b,5 pl_w,6 pl_b,
    //          7+6j:{dw,dwb,pw,pwb,xdw,xdb}
    const int G1 = 1, G2 = 32;
    const float* x_in = (const float*)d_in[0];
    float* out = (float*)d_out;

    // --- workspace carve-up ---
    char* ws = (char*)d_ws;
    float*  X1    = (float*)(ws + 0);                      // [BN,64] f32
    float*  X2    = (float*)(ws + (16u << 20));            // [BN,64] f32
    bf16_t* ABF   = (bf16_t*)(ws + (32u << 20));           // [BN,256] bf16 staging
    bf16_t* A2    = (bf16_t*)(ws + (64u << 20));           // pf output
    bf16_t* LKA   = (bf16_t*)(ws + (96u << 20));
    bf16_t* XK    = (bf16_t*)(ws + (128u << 20));
    bf16_t* T     = (bf16_t*)(ws + (160u << 20));
    float*  QGKV  = (float*)(ws + (192u << 20));           // [BN,256] f32
    float*  QSIM  = (float*)(ws + (256u << 20));           // [B*H*N,32] f32
    float*  VD    = (float*)(ws + (292u << 20));           // [BN,64] f32
    float*  KVRED = (float*)(ws + (310u << 20));           // [64,544] f32
    bf16_t* WAR   = (bf16_t*)(ws + (312u << 20));          // bf16 weight arena

    bf16_t* Wpf1 = WAR;            bf16_t* Wpw1 = WAR + 16384; bf16_t* Wpl1 = WAR + 32768;
    bf16_t* Wpf2 = WAR + 49152;    bf16_t* Wpw2 = WAR + 65536; bf16_t* Wpl2 = WAR + 81920;
    bf16_t* Wqgkv = WAR + 98304;   bf16_t* Wproj = WAR + 114688;

    auto cvt = [&](int idx, bf16_t* dst, int n) {
        f2bf_kernel<<<(n + 255) / 256, 256, 0, stream>>>((const float*)d_in[idx], dst, n);
    };
    cvt(G1 + 3, Wpf1, 256 * 64);  cvt(G1 + 5, Wpl1, 64 * 256);
    cvt(G2 + 3, Wpf2, 256 * 64);  cvt(G2 + 5, Wpl2, 64 * 256);
    for (int j = 0; j < 4; ++j) { cvt(G1 + 7 + 6 * j + 2, Wpw1 + j * 4096, 4096);
                                  cvt(G2 + 7 + 6 * j + 2, Wpw2 + j * 4096, 4096); }
    cvt(63, Wqgkv, 128 * 64);     cvt(64, Wqgkv + 8192, 128 * 64);
    cvt(65, Wproj, 64 * 64);

    auto nblk = [](int M, int O) { return (((M >> 6) * (O >> 4)) + 7) >> 3; };

    static const int KS[4] = {3, 11, 23, 31};
    auto run_gcgfn = [&](const float* xin, float* xout, int gb,
                         const bf16_t* Wpf, const bf16_t* Wpw, const bf16_t* Wpl) {
        auto gp = [&](int k) { return (const float*)d_in[gb + k]; };
        // channel-LN (eps 1e-6) -> bf16 A
        ln_c64_kernel<<<BN_ / 8, 256, 0, stream>>>(xin, gp(0), gp(1), ABF, nullptr, 1e-6f, BN_);
        // pf: [BN,64]x[64,256] -> bf16 A2
        wmma_gemm_kernel<2, 0, 64, 256><<<nblk(BN_, 256), 256, 0, stream>>>(
            ABF, Wpf, gp(4), BN_, 256, nullptr, nullptr, nullptr, nullptr, A2);
        // depthwise convs (both lka and x branches), TDM-staged LDS tiles
        for (int j = 0; j < 4; ++j)
            gcgfn_dwconv_kernel<<<B_ * (N_ / 32), 256, 0, stream>>>(
                A2, gp(7 + 6 * j), gp(7 + 6 * j + 1), gp(7 + 6 * j + 4), gp(7 + 6 * j + 5),
                KS[j], j, LKA, XK);
        // per-branch pw GEMM with fused *xk gate
        for (int j = 0; j < 4; ++j)
            wmma_gemm_kernel<2, 1, 256, 256><<<nblk(BN_, 64), 256, 0, stream>>>(
                LKA + 64 * j, Wpw + j * 4096, gp(7 + 6 * j + 3), BN_, 64,
                XK + 64 * j, nullptr, nullptr, nullptr, T + 64 * j);
        // pl GEMM with fused *scale + residual
        wmma_gemm_kernel<8, 2, 256, 64><<<nblk(BN_, 64), 256, 0, stream>>>(
            T, Wpl, gp(6), BN_, 64, nullptr, gp(2), xin, xout, nullptr);
    };

    // ---------------- forward ----------------
    run_gcgfn(x_in, X1, G1, Wpf1, Wpw1, Wpl1);

    // norm1 (eps 1e-5): bf16 for GEMM + f32 copy (residual base for attention)
    ln_c64_kernel<<<BN_ / 8, 256, 0, stream>>>(X1, (const float*)d_in[72],
        (const float*)d_in[73], ABF, X2, 1e-5f, BN_);

    // fused q|g|k|v GEMM: [BN,64] x [64,256]
    wmma_gemm_kernel<2, 3, 64, 256><<<nblk(BN_, 256), 256, 0, stream>>>(
        ABF, Wqgkv, nullptr, BN_, 256, nullptr, nullptr, nullptr, QGKV, nullptr);

    zero_f32_kernel<<<(64 * 544 + 255) / 256, 256, 0, stream>>>(KVRED, 64 * 544);
    attn_feat_kernel<<<dim3(B_ * H_, N_ / 256), 256, 0, stream>>>(
        QGKV, (const float*)d_in[70], (const float*)d_in[69], (const float*)d_in[71],
        QSIM, KVRED);
    attn_dwc_kernel<<<(B_ * 16 * N_ * 4) / 256, 256, 0, stream>>>(
        QGKV, (const float*)d_in[67], (const float*)d_in[68], VD);
    attn_out_kernel<<<dim3(B_ * H_, N_ / 256), 256, 0, stream>>>(
        QSIM, KVRED, VD, QGKV, ABF);

    // proj GEMM + residual (residual base = norm1 output X2)
    wmma_gemm_kernel<2, 4, 64, 64><<<nblk(BN_, 64), 256, 0, stream>>>(
        ABF, Wproj, (const float*)d_in[66], BN_, 64, nullptr, nullptr, X2, X1, nullptr);

    run_gcgfn(X1, X2, G2, Wpf2, Wpw2, Wpl2);

    biasnorm_kernel<<<BN_ / 8, 256, 0, stream>>>(X2, (const float*)d_in[74],
        (const float*)d_in[75], out, BN_);
}
